// AttentionTSSA_84112639525328
// MI455X (gfx1250) — compile-verified, matrix operand order checked by reading, not run on hardware
//
#include <hip/hip_runtime.h>
#include <hip/hip_bf16.h>

// Problem constants from the reference: B=8, L=N=1024, H=8, E=d=64
#define NB   8
#define NH   8
#define NN   1024
#define ND   64
#define BH   (NB*NH)               // 64
#define OUT0_ELEMS (NB*NN*NH*ND)   // 4,194,304 floats (first output "out")

typedef _Float16 v16h __attribute__((ext_vector_type(16)));
typedef _Float16 v8h  __attribute__((ext_vector_type(8)));
typedef float    v8f  __attribute__((ext_vector_type(8)));

// ---------------------------------------------------------------------------
// Kernel 1: per (b,h) block.
//   Phase A: colsumsq over tokens -> scale[d] = 1/max(sqrt(cs),eps)^2
//   Phase B: per-row: energy[b,h,n] = sum_d w^2*scale[d];
//            w_unit (f16) = w / max(||row||,eps)
// ---------------------------------------------------------------------------
__global__ void k1_norms(const float* __restrict__ q,
                         float* __restrict__ energy,
                         _Float16* __restrict__ wU) {
    const int bh = blockIdx.x;
    const int b = bh >> 3, h = bh & 7;
    // q index: ((b*NN + n)*NH + h)*ND + d  ->  base + n*(NH*ND) + d
    const float* base = q + (size_t)b * (NN * NH * ND) + (size_t)h * ND;

    __shared__ float lds[256];
    __shared__ float scale[ND];

    const int t = threadIdx.x;
    const int d = t & 63, noff = t >> 6;

    // Phase A: column sum of squares (over tokens)
    float acc = 0.0f;
    for (int n = noff; n < NN; n += 4) {
        float v = base[n * (NH * ND) + d];
        acc += v * v;
    }
    lds[t] = acc;
    __syncthreads();
    if (t < ND) {
        float cs = lds[t] + lds[t + 64] + lds[t + 128] + lds[t + 192];
        float cn = fmaxf(sqrtf(cs), 1e-12f);
        scale[t] = 1.0f / (cn * cn);
    }
    __syncthreads();

    // Phase B: one wave per row (8 waves, rows strided by 8)
    const int lane = t & 31, wave = t >> 5;
    const float s0 = scale[2 * lane], s1 = scale[2 * lane + 1];
    for (int n = wave; n < NN; n += 8) {
        const float2 wv = *(const float2*)(base + n * (NH * ND) + 2 * lane);
        float sq = wv.x * wv.x + wv.y * wv.y;       // row ||.||^2 partial
        float ep = wv.x * wv.x * s0 + wv.y * wv.y * s1; // energy partial
        float rs = sq, re = ep;
        #pragma unroll
        for (int m = 16; m; m >>= 1) {
            rs += __shfl_xor(rs, m, 32);
            re += __shfl_xor(re, m, 32);
        }
        if (lane == 0) energy[bh * NN + n] = re;
        float inv = 1.0f / fmaxf(sqrtf(rs), 1e-12f);
        union { _Float16 f[2]; unsigned int u; } p;
        p.f[0] = (_Float16)(wv.x * inv);
        p.f[1] = (_Float16)(wv.y * inv);
        *(unsigned int*)(&wU[(size_t)bh * (NN * ND) + (size_t)n * ND + 2 * lane]) = p.u;
    }
}

// ---------------------------------------------------------------------------
// Kernel 2: softmax over the HEAD dim (8 values) per (b,n)
// ---------------------------------------------------------------------------
__global__ void k2_softmax(const float* __restrict__ energy,
                           const float* __restrict__ temp,
                           float* __restrict__ Pi) {
    const int idx = blockIdx.x * blockDim.x + threadIdx.x; // 0..8191
    const int b = idx >> 10, n = idx & (NN - 1);
    float e[NH];
    float mx = -3.402823466e38f;
    #pragma unroll
    for (int h = 0; h < NH; ++h) {
        e[h] = energy[(b * NH + h) * NN + n] * temp[h];
        mx = fmaxf(mx, e[h]);
    }
    float sum = 0.0f;
    #pragma unroll
    for (int h = 0; h < NH; ++h) { e[h] = expf(e[h] - mx); sum += e[h]; }
    float rinv = 1.0f / sum;
    #pragma unroll
    for (int h = 0; h < NH; ++h) Pi[(b * NH + h) * NN + n] = e[h] * rinv;
}

// ---------------------------------------------------------------------------
// Kernel 3: per (b,h): S=sum Pi; dots[d]=sum Pi_norm*w^2; attn=1/(1+dots);
//           out[b,n,h,d] = -(w*Pi)*attn[d]
// ---------------------------------------------------------------------------
__global__ void k3_out(const float* __restrict__ q,
                       const float* __restrict__ Pi,
                       float* __restrict__ out) {
    const int bh = blockIdx.x;
    const int b = bh >> 3, h = bh & 7;
    const float* base = q + (size_t)b * (NN * NH * ND) + (size_t)h * ND;
    float* obase = out + (size_t)b * (NN * NH * ND) + (size_t)h * ND;
    const float* pi = Pi + bh * NN;

    __shared__ float lds[256];
    __shared__ float attnv[ND];

    const int t = threadIdx.x;
    const int d = t & 63, noff = t >> 6;

    // S = sum_n Pi
    float ps = 0.0f;
    for (int n = t; n < NN; n += 256) ps += pi[n];
    lds[t] = ps;
    __syncthreads();
    #pragma unroll
    for (int s = 128; s; s >>= 1) {
        if (t < s) lds[t] += lds[t + s];
        __syncthreads();
    }
    const float S = lds[0];
    __syncthreads();

    // dots[d]
    float pd = 0.0f;
    for (int n = noff; n < NN; n += 4) {
        float v = base[n * (NH * ND) + d];
        pd += pi[n] * v * v;
    }
    lds[t] = pd;
    __syncthreads();
    if (t < ND) {
        float dots = (lds[t] + lds[t + 64] + lds[t + 128] + lds[t + 192]) / (S + 1e-8f);
        attnv[t] = 1.0f / (1.0f + dots);
    }
    __syncthreads();

    const float av = attnv[d];
    for (int n = noff; n < NN; n += 4) {
        float v = base[n * (NH * ND) + d];
        obase[n * (NH * ND) + d] = -(v * pi[n]) * av;
    }
}

// ---------------------------------------------------------------------------
// Kernel 4: WMMA gram matrix. One wave32 per 16x16 tile of G = U U^T,
// two v_wmma_f32_16x16x32_f16 per tile (K = 64).
// 16-bit A layout: lane<16 -> row M=lane, K {kb+0..7, kb+16..23};
//                  lane>=16 -> row M=lane-16, K {kb+8..15, kb+24..31}.
// B (32x16) layout is identical with M<->N, so the same loader serves both.
// ---------------------------------------------------------------------------
__device__ __forceinline__ v16h load_tile(const _Float16* __restrict__ U,
                                          int row0, int kb, int lane) {
    const int m = lane & 15, hi = lane >> 4;
    const _Float16* p = U + (size_t)(row0 + m) * ND + kb + hi * 8;
    v8h a0 = *(const v8h*)(p);
    v8h a1 = *(const v8h*)(p + 16);
    v16h r;
    #pragma unroll
    for (int i = 0; i < 8; ++i) { r[i] = a0[i]; r[8 + i] = a1[i]; }
    return r;
}

__global__ void k4_gram(const _Float16* __restrict__ wU,
                        float* __restrict__ attnOut) {
    const int wid  = (blockIdx.x * blockDim.x + threadIdx.x) >> 5;
    const int lane = threadIdx.x & 31;

    // tiles: bh(64) x ti(64) x tj(64)
    const int bh  = wid >> 12;
    const int rem = wid & 4095;
    const int ti  = rem >> 6, tj = rem & 63;

    const _Float16* U = wU + (size_t)bh * (NN * ND);

    v8f c = {};
    v16h a0 = load_tile(U, ti * 16, 0, lane);
    v16h b0 = load_tile(U, tj * 16, 0, lane);
    c = __builtin_amdgcn_wmma_f32_16x16x32_f16(false, a0, false, b0,
                                               (short)0, c, false, false);
    v16h a1 = load_tile(U, ti * 16, 32, lane);
    v16h b1 = load_tile(U, tj * 16, 32, lane);
    c = __builtin_amdgcn_wmma_f32_16x16x32_f16(false, a1, false, b1,
                                               (short)0, c, false, false);

    const int b = bh >> 3, h = bh & 7;
    // attn_reshaped idx: ((b*NN + n)*NH + h)*NN + m
    const int n0 = ti * 16 + 8 * (lane >> 4);
    const int m0 = tj * 16 + (lane & 15);
    float* obase = attnOut + (size_t)b * (NN * NH * NN) + (size_t)h * NN;
    #pragma unroll
    for (int r = 0; r < 8; ++r) {
        size_t n = (size_t)(n0 + r);
        obase[n * (NH * NN) + m0] = (c[r] + 1.0f) * 0.5f;
    }
}

// ---------------------------------------------------------------------------
extern "C" void kernel_launch(void* const* d_in, const int* in_sizes, int n_in,
                              void* d_out, int out_size, void* d_ws, size_t ws_size,
                              hipStream_t stream) {
    (void)in_sizes; (void)n_in; (void)out_size; (void)ws_size;
    const float* q    = (const float*)d_in[0];   // [8,1024,8,64] f32
    const float* temp = (const float*)d_in[1];   // [8,1] f32

    float* out     = (float*)d_out;              // first output: 4,194,304 floats
    float* attnOut = out + OUT0_ELEMS;           // second output: 67,108,864 floats

    // Workspace layout
    _Float16* wU   = (_Float16*)d_ws;                               // 8 MB
    float* energy  = (float*)((char*)d_ws + (size_t)BH * NN * ND * 2);      // 256 KB
    float* Pi      = energy + BH * NN;                               // 256 KB

    k1_norms  <<<BH,    256, 0, stream>>>(q, energy, wU);
    k2_softmax<<<NB * NN / 256, 256, 0, stream>>>(energy, temp, Pi);
    k3_out    <<<BH,    256, 0, stream>>>(q, Pi, out);
    // 64 (bh) * 64 * 64 tiles, 1 wave per tile, 8 waves per block
    k4_gram   <<<BH * 64 * 64 / 8, 256, 0, stream>>>(wU, attnOut);
}
